// PointPairNet_73547019976965
// MI455X (gfx1250) — matrix operand order repaired and checked by reading, not compiled, not run on hardware
//
#include <hip/hip_runtime.h>

// ---------------------------------------------------------------------------
// PointPairNet fused kernel for MI455X (gfx1250, wave32, WMMA).
//
// Math: per batch b (B=4, N=512, C1=64, C2=64, C3=128):
//   H1[i,j,:] = relu(Ai[i,:] + Aj[j,:])          (b1 folded into Ai)
//   H2        = relu(H1 @ W2^T + b2)             (b2 folded into WMMA C operand)
//   H3        = H2 @ W3^T + b3                   (b3 folded into WMMA C operand)
//   P[b,:]    = max over (i,j) of H3
//   Y         = relu(P @ F1^T + bf1) @ F2^T + bf2
//
// Layer2/3 run on v_wmma_f32_16x16x32_f16 (fp16 in, fp32 accumulate).
// Weights + Ai tile are staged to LDS with global_load_async_to_lds_b128.
// ---------------------------------------------------------------------------

typedef __attribute__((ext_vector_type(16))) _Float16 v16h;
typedef __attribute__((ext_vector_type(8)))  float    v8f;

#define B_   4
#define N_   512
#define C1_  64
#define C3_  128
#define NPAD 72   // LDS row pitch in halves (144B, 16B aligned, breaks bank alignment)

// Wave-private LDS producer->consumer ordering (no cross-wave sharing in the
// hot loop, so a split-counter wait is enough; no s_barrier needed).
#define WAVE_LDS_SYNC() do {                                   \
    __asm__ volatile("s_wait_dscnt 0" ::: "memory");           \
    __builtin_amdgcn_wave_barrier();                           \
  } while (0)

// Async global->LDS copy of one 16B chunk per lane (ASYNCcnt tracked).
// dst = raw LDS byte offset (low 32 bits of shared-aperture flat address),
// src = per-lane byte offset, base = 64-bit global base (SGPR pair).
#define ASYNC_LDS_B128(dst, src, base)                                  \
  __asm__ volatile("global_load_async_to_lds_b128 %0, %1, %2"           \
                   :: "v"(dst), "v"(src), "s"(base) : "memory")

union Frag16 {                 // one WMMA 16x16x32 fp16 operand (8 VGPRs)
  v16h  v;
  uint4 q[2];
};

union H8 {                     // 8 packed halves for a b128 LDS store
  _Float16 h[8];
  uint4    q;
};

union HU { unsigned short u; _Float16 h; };

__device__ inline float atomicMaxF(float* addr, float val) {
  unsigned int* ua = reinterpret_cast<unsigned int*>(addr);
  unsigned int old = *ua;
  while (true) {
    float cur = __uint_as_float(old);
    if (cur >= val) break;
    unsigned int assumed = old;
    old = atomicCAS(ua, assumed, __float_as_uint(val));
    if (old == assumed) break;
  }
  return __uint_as_float(old);
}

// ---------------------------------------------------------------------------
// Kernel 0: Ai/Aj prologue.  Aj[n,c] = X[n,:]·W1[c,0:3];
//                            Ai[n,c] = X[n,:]·W1[c,3:6] + b1[c].
// ---------------------------------------------------------------------------
__global__ void prep_aiaj(const float* __restrict__ X,
                          const float* __restrict__ W1,
                          const float* __restrict__ b1,
                          float* __restrict__ Ai,
                          float* __restrict__ Aj) {
  int idx = blockIdx.x * blockDim.x + threadIdx.x;     // flat (b*n)
  if (idx >= B_ * N_) return;
  const float x0 = X[idx * 3 + 0];
  const float x1 = X[idx * 3 + 1];
  const float x2 = X[idx * 3 + 2];
  #pragma unroll 4
  for (int c = 0; c < C1_; ++c) {
    const float* w = W1 + c * 6;
    Aj[idx * C1_ + c] = w[0] * x0 + w[1] * x1 + w[2] * x2;
    Ai[idx * C1_ + c] = w[3] * x0 + w[4] * x1 + w[5] * x2 + b1[c];
  }
}

// ---------------------------------------------------------------------------
// Kernel 1: convert W2/W3 to fp16 and init pooled-max buffer to -inf.
// ---------------------------------------------------------------------------
__global__ void prep_wp(const float* __restrict__ W2,
                        const float* __restrict__ W3,
                        unsigned short* __restrict__ W2h,
                        unsigned short* __restrict__ W3h,
                        float* __restrict__ P) {
  int t = threadIdx.x;                                  // one block, 256 threads
  for (int k = t; k < 64 * 64; k += 256)  { HU u; u.h = (_Float16)W2[k]; W2h[k] = u.u; }
  for (int k = t; k < 128 * 64; k += 256) { HU u; u.h = (_Float16)W3[k]; W3h[k] = u.u; }
  for (int k = t; k < B_ * C3_; k += 256) P[k] = -INFINITY;
}

// ---------------------------------------------------------------------------
// Kernel 2: fused pair MLP + max pool.
// grid = (N/16, N/128, B), block = 256 (8 waves).
// Wave w owns j in [blockIdx.y*128 + 16w, +16) and loops over 16 i values.
// Hot loop is VMEM-free: Aj row + biases in registers, Ai tile + weights in LDS.
// ---------------------------------------------------------------------------
__global__ __launch_bounds__(256)
void pairnet_main(const float* __restrict__ Ai,
                  const float* __restrict__ Aj,
                  const unsigned short* __restrict__ W2h,
                  const unsigned short* __restrict__ W3h,
                  const float* __restrict__ b2,
                  const float* __restrict__ b3,
                  float* __restrict__ P) {
  __shared__ alignas(16) _Float16 sW2[64][NPAD];       //  9.0 KB  W2 [out][in]
  __shared__ alignas(16) _Float16 sW3[128][NPAD];      // 18.0 KB  W3 [out][in]
  __shared__ alignas(16) _Float16 sH1[8][16][NPAD];    // 18.0 KB  per-wave H1 tile
  __shared__ alignas(16) _Float16 sH2[8][16][NPAD];    // 18.0 KB  per-wave H2 tile
  __shared__ alignas(16) float    sAi[16][64];         //  4.0 KB  block's Ai tile

  const int tid  = threadIdx.x;
  const int wave = tid >> 5;
  const int lane = tid & 31;
  const int b    = blockIdx.z;
  const int i0   = blockIdx.x * 16;
  const int j0   = blockIdx.y * 128 + wave * 16;

  // ---- async stage: W2h (8KB), W3h (16KB), Ai tile (4KB) -> LDS ----------
  {
    const unsigned ldsW2 = (unsigned)(size_t)&sW2[0][0];
    const unsigned ldsW3 = (unsigned)(size_t)&sW3[0][0];
    const unsigned ldsAi = (unsigned)(size_t)&sAi[0][0];
    const unsigned long long gW2 = (unsigned long long)(size_t)W2h;
    const unsigned long long gW3 = (unsigned long long)(size_t)W3h;
    const unsigned long long gAi =
        (unsigned long long)(size_t)(Ai + ((size_t)b * N_ + i0) * C1_);
    // W2: 512 16B chunks, 8 per 128B row -> padded rows in LDS
    #pragma unroll
    for (int s = 0; s < 2; ++s) {
      int q = tid + s * 256;
      unsigned dst = ldsW2 + (unsigned)((q >> 3) * (NPAD * 2) + (q & 7) * 16);
      ASYNC_LDS_B128(dst, (unsigned)(q * 16), gW2);
    }
    // W3: 1024 16B chunks
    #pragma unroll
    for (int s = 0; s < 4; ++s) {
      int q = tid + s * 256;
      unsigned dst = ldsW3 + (unsigned)((q >> 3) * (NPAD * 2) + (q & 7) * 16);
      ASYNC_LDS_B128(dst, (unsigned)(q * 16), gW3);
    }
    // Ai tile: 256 16B chunks, unpadded (reads are lane-uniform broadcasts)
    {
      unsigned dst = ldsAi + (unsigned)(tid * 16);
      ASYNC_LDS_B128(dst, (unsigned)(tid * 16), gAi);
    }
    __asm__ volatile("s_wait_asynccnt 0" ::: "memory");
  }
  __syncthreads();

  const int mrow = lane & 15;     // pair row inside the 16-row M tile (= local j)
  const int half = lane >> 4;     // lane half selects K-interleave / M-offset
  const int cb   = half * 32;     // this lane builds channels [cb, cb+32) of H1

  // ---- loop-invariant register hoists ------------------------------------
  float ajr[32];                  // Aj[j0+mrow, cb..cb+32) — invariant over i
  {
    const float* AjRow = Aj + ((size_t)b * N_ + (j0 + mrow)) * C1_ + cb;
    #pragma unroll
    for (int c = 0; c < 32; c += 4)
      *(float4*)&ajr[c] = *(const float4*)(AjRow + c);
  }
  float b2r[4], b3r[8];           // per-lane biases (lane&15 == output channel)
  #pragma unroll
  for (int nt = 0; nt < 4; ++nt) b2r[nt] = b2[nt * 16 + mrow];
  #pragma unroll
  for (int nt = 0; nt < 8; ++nt) b3r[nt] = b3[nt * 16 + mrow];

  float pmax[8];
  #pragma unroll
  for (int q = 0; q < 8; ++q) pmax[q] = -INFINITY;

  for (int it = 0; it < 16; ++it) {
    // ---- build H1 tile (16 pairs x 64 ch) in fp16, fp32 math -------------
    #pragma unroll
    for (int c = 0; c < 32; c += 8) {
      float4 a0 = *(const float4*)&sAi[it][cb + c];       // lane-uniform bcast
      float4 a1 = *(const float4*)&sAi[it][cb + c + 4];
      H8 o;
      o.h[0] = (_Float16)fmaxf(a0.x + ajr[c + 0], 0.0f);
      o.h[1] = (_Float16)fmaxf(a0.y + ajr[c + 1], 0.0f);
      o.h[2] = (_Float16)fmaxf(a0.z + ajr[c + 2], 0.0f);
      o.h[3] = (_Float16)fmaxf(a0.w + ajr[c + 3], 0.0f);
      o.h[4] = (_Float16)fmaxf(a1.x + ajr[c + 4], 0.0f);
      o.h[5] = (_Float16)fmaxf(a1.y + ajr[c + 5], 0.0f);
      o.h[6] = (_Float16)fmaxf(a1.z + ajr[c + 6], 0.0f);
      o.h[7] = (_Float16)fmaxf(a1.w + ajr[c + 7], 0.0f);
      *(uint4*)&sH1[wave][mrow][cb + c] = o.q;
    }
    WAVE_LDS_SYNC();

    // ---- A fragments for layer2: 16x32 fp16, K-steps kk*32 ---------------
    // ISA layout: lanes 0-15 row M=lane hold K {0-7,16-23}; lanes 16-31 hold
    // K {8-15,24-31} -> two ds_load_b128 per fragment.
    Frag16 A1[2];
    #pragma unroll
    for (int kk = 0; kk < 2; ++kk) {
      const _Float16* base = &sH1[wave][mrow][kk * 32 + half * 8];
      A1[kk].q[0] = *(const uint4*)(base);        // V0-V3
      A1[kk].q[1] = *(const uint4*)(base + 16);   // V4-V7
    }

    // ---- layer2: [16,64] @ W2^T[64,64]; b2 folded into C operand ---------
    v8f acc2[4];
    #pragma unroll
    for (int nt = 0; nt < 4; ++nt) {
      v8f c0;
      #pragma unroll
      for (int r = 0; r < 8; ++r) c0[r] = b2r[nt];  // C[m][n]=b2[n], lane==n
      #pragma unroll
      for (int kk = 0; kk < 2; ++kk) {
        // B 32x16: lane = N column, same K interleave as A (K {0-7,16-23} /
        // {8-15,24-31} per lane half) -> two b128 LDS loads.
        Frag16 Bf;
        const _Float16* wb = &sW2[nt * 16 + mrow][kk * 32 + half * 8];
        Bf.q[0] = *(const uint4*)(wb);
        Bf.q[1] = *(const uint4*)(wb + 16);
        c0 = __builtin_amdgcn_wmma_f32_16x16x32_f16(
            false, A1[kk].v, false, Bf.v, (short)0, c0, false, false);
      }
      acc2[nt] = c0;
    }

    // ---- relu, scatter H2 to LDS (C/D layout: M = r + 8*half) ------------
    #pragma unroll
    for (int nt = 0; nt < 4; ++nt) {
      const int n = nt * 16 + mrow;
      #pragma unroll
      for (int r = 0; r < 8; ++r) {
        sH2[wave][r + half * 8][n] = (_Float16)fmaxf(acc2[nt][r], 0.0f);
      }
    }
    WAVE_LDS_SYNC();

    Frag16 A2[2];
    #pragma unroll
    for (int kk = 0; kk < 2; ++kk) {
      const _Float16* base = &sH2[wave][mrow][kk * 32 + half * 8];
      A2[kk].q[0] = *(const uint4*)(base);
      A2[kk].q[1] = *(const uint4*)(base + 16);
    }

    // ---- layer3: [16,64] @ W3^T[64,128]; b3 folded into C operand,
    //      max-fold accumulators in registers --------------------------------
    #pragma unroll
    for (int nt = 0; nt < 8; ++nt) {
      v8f acc;
      #pragma unroll
      for (int r = 0; r < 8; ++r) acc[r] = b3r[nt];
      #pragma unroll
      for (int kk = 0; kk < 2; ++kk) {
        Frag16 Bf;
        const _Float16* wb = &sW3[nt * 16 + mrow][kk * 32 + half * 8];
        Bf.q[0] = *(const uint4*)(wb);
        Bf.q[1] = *(const uint4*)(wb + 16);
        acc = __builtin_amdgcn_wmma_f32_16x16x32_f16(
            false, A2[kk].v, false, Bf.v, (short)0, acc, false, false);
      }
      float m01 = fmaxf(fmaxf(acc[0], acc[1]), fmaxf(acc[2], acc[3]));
      float m23 = fmaxf(fmaxf(acc[4], acc[5]), fmaxf(acc[6], acc[7]));
      pmax[nt] = fmaxf(pmax[nt], fmaxf(m01, m23));
    }
  }

  // ---- pool epilogue: combine lane halves (same output channel d lives in
  // lanes l and l^16), then CAS-max into global P[b, d] ---------------------
  #pragma unroll
  for (int nt = 0; nt < 8; ++nt) {
    float other = __shfl_xor(pmax[nt], 16, 32);
    float v = fmaxf(pmax[nt], other);
    if (half == 0) atomicMaxF(&P[b * C3_ + nt * 16 + mrow], v);
  }
}

// ---------------------------------------------------------------------------
// Kernel 3: tiny head.  Y = relu(P @ F1^T + bf1) @ F2^T + bf2.
// (b3 already folded into P by the main kernel.)  One block per batch.
// ---------------------------------------------------------------------------
__global__ void head_kernel(const float* __restrict__ P,
                            const float* __restrict__ F1,
                            const float* __restrict__ bf1,
                            const float* __restrict__ F2,
                            const float* __restrict__ bf2,
                            float* __restrict__ Y) {
  __shared__ float sp[128];
  __shared__ float sh[64];
  const int b = blockIdx.x;
  const int t = threadIdx.x;
  sp[t] = P[b * C3_ + t];
  __syncthreads();
  if (t < 64) {
    float s = bf1[t];
    #pragma unroll 8
    for (int k = 0; k < 128; ++k) s += F1[t * 128 + k] * sp[k];
    sh[t] = fmaxf(s, 0.0f);
  }
  __syncthreads();
  if (t < 2) {
    float s = bf2[t];
    #pragma unroll 8
    for (int k = 0; k < 64; ++k) s += F2[t * 64 + k] * sh[k];
    Y[b * 2 + t] = s;
  }
}

// ---------------------------------------------------------------------------
// Host launcher.
// Inputs (setup_inputs order, all fp32):
//   0:X 1:W1 2:b1 3:W2 4:b2 5:W3 6:b3 7:F1 8:bf1 9:F2 10:bf2
// Output: Y [4,2] fp32.
// ---------------------------------------------------------------------------
extern "C" void kernel_launch(void* const* d_in, const int* in_sizes, int n_in,
                              void* d_out, int out_size, void* d_ws, size_t ws_size,
                              hipStream_t stream) {
  const float* X   = (const float*)d_in[0];
  const float* W1  = (const float*)d_in[1];
  const float* b1  = (const float*)d_in[2];
  const float* W2  = (const float*)d_in[3];
  const float* b2  = (const float*)d_in[4];
  const float* W3  = (const float*)d_in[5];
  const float* b3  = (const float*)d_in[6];
  const float* F1  = (const float*)d_in[7];
  const float* bf1 = (const float*)d_in[8];
  const float* F2  = (const float*)d_in[9];
  const float* bf2 = (const float*)d_in[10];
  float* Y = (float*)d_out;

  // Workspace carve-up (all 16B aligned):
  char* ws = (char*)d_ws;
  float*          Aj  = (float*)(ws);                                // 512 KB
  float*          Ai  = (float*)(ws + (size_t)B_ * N_ * C1_ * 4);    // 512 KB
  unsigned short* W2h = (unsigned short*)(ws + (size_t)2 * B_ * N_ * C1_ * 4);
  unsigned short* W3h = W2h + 64 * 64;
  float*          P   = (float*)((char*)(W3h + 128 * 64));           // [4,128]

  prep_aiaj<<<(B_ * N_ + 255) / 256, 256, 0, stream>>>(X, W1, b1, Ai, Aj);
  prep_wp<<<1, 256, 0, stream>>>(W2, W3, W2h, W3h, P);
  pairnet_main<<<dim3(N_ / 16, N_ / 128, B_), 256, 0, stream>>>(
      Ai, Aj, W2h, W3h, b2, b3, P);
  head_kernel<<<B_, 128, 0, stream>>>(P, F1, bf1, F2, bf2, Y);
}
